// GraphEmbeddings_32933809226087
// MI455X (gfx1250) — matrix operand order repaired and checked by reading, not compiled, not run on hardware
//
#include <hip/hip_runtime.h>
#include <math.h>

typedef __attribute__((ext_vector_type(16))) _Float16 v16h;
typedef __attribute__((ext_vector_type(2)))  _Float16 h2;
typedef __attribute__((ext_vector_type(8)))  float    v8f;
typedef __attribute__((ext_vector_type(4)))  float    f4;

// ---------------------------------------------------------------------------
// Pass 0: init per-node accumulators (must happen every call: ws is not
// re-poisoned between replays) and zero the edge-attr sum.
// ---------------------------------------------------------------------------
__global__ void k_init(unsigned* __restrict__ nodemax,
                       float* __restrict__ denom,
                       float* __restrict__ wsum,
                       float* __restrict__ easum, int Nn) {
    int i = blockIdx.x * blockDim.x + threadIdx.x;
    if (i < Nn) {
        nodemax[i] = 0u;     // encoded -NaN; every node has a self-loop so it's overwritten
        denom[i]   = 0.f;
        wsum[i]    = 0.f;
    }
    if (i == 0) easum[0] = 0.f;
}

// ---------------------------------------------------------------------------
// Pass 0b: sum(edge_attr) for the self-loop fill value (mean).
// ---------------------------------------------------------------------------
__global__ void k_easum(const float* __restrict__ ea, float* __restrict__ easum, int E) {
    __shared__ float red[256];
    float s = 0.f;
    for (int i = blockIdx.x * blockDim.x + threadIdx.x; i < E;
         i += gridDim.x * blockDim.x)
        s += ea[i];
    red[threadIdx.x] = s;
    __syncthreads();
    for (int w = 128; w > 0; w >>= 1) {
        if (threadIdx.x < w) red[threadIdx.x] += red[threadIdx.x + w];
        __syncthreads();
    }
    if (threadIdx.x == 0) atomicAdd(easum, red[0]);
}

// ---------------------------------------------------------------------------
// Pass 1: per-edge attention logits via WMMA.
// One wave handles 16 edges per tile:
//   A[m][k] = leaky_relu(x[src_m]*Wl[k] + x[dst_m]*Wr[k] + ea_m*We[k] + bl[k]+br[k])
//   B[k][n] = att[k]  (broadcast across the 16 columns)
//   D[m][*] = logit of edge m  (f32 accumulate), two K=32 chunks cover C=64.
// A built with packed f16 FMAs; lrelu(u) = max(u, 0.2*u).
// The logits buffer is padded to a multiple of 16 so the accumulator can be
// written back unconditionally as two b128 stores (no per-element predication).
// ---------------------------------------------------------------------------
__global__ void k_logits(const float* __restrict__ x,
                         const int*   __restrict__ esrc,
                         const int*   __restrict__ edst,
                         const float* __restrict__ ea,
                         const float* __restrict__ Wl, const float* __restrict__ bl,
                         const float* __restrict__ Wr, const float* __restrict__ br,
                         const float* __restrict__ We, const float* __restrict__ att,
                         const float* __restrict__ easum,
                         float* __restrict__ logits,
                         int E, int Etot, int ntiles) {
    const int lane   = threadIdx.x & 31;
    const int wid    = blockIdx.x * (blockDim.x >> 5) + (threadIdx.x >> 5);
    const int nwaves = gridDim.x * (blockDim.x >> 5);
    const int half   = lane >> 4;   // which 16-lane half of the wave
    const int m      = lane & 15;   // edge row within the 16-edge tile

    // Per-lane packed constants following the 16-bit A (16x32) VGPR layout:
    //   VGPR p, lanes 0-15 : K = 2*(p&3) + 16*(p>>2)
    //   VGPR p, lanes 16-31: K = above + 8
    // and the same K mapping for the 16-bit B (32x16) operand.
    h2 wl2[2][8], wr2[2][8], we2[2][8], bb2[2][8];
    v16h Bq[2];
#pragma unroll
    for (int q = 0; q < 2; ++q) {
#pragma unroll
        for (int p = 0; p < 8; ++p) {
            int c = 32 * q + ((p & 3) << 1) + ((p >> 2) << 4) + (half << 3);
            wl2[q][p] = h2{(_Float16)Wl[c], (_Float16)Wl[c + 1]};
            wr2[q][p] = h2{(_Float16)Wr[c], (_Float16)Wr[c + 1]};
            we2[q][p] = h2{(_Float16)We[c], (_Float16)We[c + 1]};
            bb2[q][p] = h2{(_Float16)(bl[c] + br[c]),
                           (_Float16)(bl[c + 1] + br[c + 1])};
            int kb = 32 * q + ((p & 3) << 1) + ((p >> 2) << 4) + (half << 3);
            Bq[q][2 * p]     = (_Float16)att[kb];
            Bq[q][2 * p + 1] = (_Float16)att[kb + 1];
        }
    }
    const float mean = easum[0] / (float)E;
    const _Float16 slope = (_Float16)0.2f;

    for (int tile = wid; tile < ntiles; tile += nwaves) {
        int e = tile * 16 + m;
        float a = 0.f, b = 0.f, ev = 0.f;
        if (e < Etot) {
            int s, d;
            if (e < E) { s = esrc[e]; d = edst[e]; ev = ea[e]; }
            else       { s = e - E;   d = s;       ev = mean;  }
            a = x[s];
            b = x[d];
        }
        h2 a2 = h2{(_Float16)a,  (_Float16)a};
        h2 b2 = h2{(_Float16)b,  (_Float16)b};
        h2 e2 = h2{(_Float16)ev, (_Float16)ev};

        v8f acc = {};
#pragma unroll
        for (int q = 0; q < 2; ++q) {
            v16h A;
#pragma unroll
            for (int p = 0; p < 8; ++p) {
                h2 u = a2 * wl2[q][p] + b2 * wr2[q][p] + e2 * we2[q][p] + bb2[q][p];
                h2 r = __builtin_elementwise_max(u, u * slope);  // leaky_relu
                A[2 * p]     = r.x;
                A[2 * p + 1] = r.y;
            }
            acc = __builtin_amdgcn_wmma_f32_16x16x32_f16(
                false, A, false, Bq[q], (short)0, acc, false, false);
        }
        // D layout: VGPR r, lanes 0-15 -> M=r; lanes 16-31 -> M=r+8.
        // All 16 columns carry identical values; lane 0 / lane 16 write them out
        // unconditionally as two b128 stores (buffer is padded to tile size).
        if (m == 0) {
            f4* p = (f4*)(logits + tile * 16 + half * 8);
            p[0] = f4{acc[0], acc[1], acc[2], acc[3]};
            p[1] = f4{acc[4], acc[5], acc[6], acc[7]};
        }
    }
}

// Order-preserving float<->uint mapping so atomicMax works on floats.
__device__ __forceinline__ unsigned enc_f(float f) {
    unsigned u = __float_as_uint(f);
    return (u & 0x80000000u) ? ~u : (u | 0x80000000u);
}
__device__ __forceinline__ float dec_f(unsigned e) {
    unsigned u = (e & 0x80000000u) ? (e & 0x7FFFFFFFu) : ~e;
    return __uint_as_float(u);
}

// ---------------------------------------------------------------------------
// Pass 2: segment max of logits over destination nodes.
// ---------------------------------------------------------------------------
__global__ void k_segmax(const float* __restrict__ logits,
                         const int* __restrict__ edst,
                         unsigned* __restrict__ nodemax, int E, int Etot) {
    int e = blockIdx.x * blockDim.x + threadIdx.x;
    if (e >= Etot) return;
    int d = (e < E) ? edst[e] : (e - E);
    atomicMax(&nodemax[d], enc_f(logits[e]));
}

// ---------------------------------------------------------------------------
// Pass 3: exp(logit - max) accumulation: denom and x[src]-weighted sum.
// ---------------------------------------------------------------------------
__global__ void k_expsum(const float* __restrict__ logits,
                         const int* __restrict__ esrc,
                         const int* __restrict__ edst,
                         const float* __restrict__ x,
                         const unsigned* __restrict__ nodemax,
                         float* __restrict__ denom, float* __restrict__ wsum,
                         int E, int Etot) {
    int e = blockIdx.x * blockDim.x + threadIdx.x;
    if (e >= Etot) return;
    int s, d;
    if (e < E) { s = esrc[e]; d = edst[e]; }
    else       { s = e - E;   d = s;       }
    float ex = __expf(logits[e] - dec_f(nodemax[d]));
    atomicAdd(&denom[d], ex);
    atomicAdd(&wsum[d],  ex * x[s]);
}

// ---------------------------------------------------------------------------
// Pass 4: rank-1 finalize: out[n,c] = bias[c] + bl[c] + Wl[c] * (wsum/denom).
// ---------------------------------------------------------------------------
__global__ void k_out(const float* __restrict__ Wl, const float* __restrict__ bl,
                      const float* __restrict__ bias,
                      const float* __restrict__ denom, const float* __restrict__ wsum,
                      float* __restrict__ out, int total, int C) {
    int i = blockIdx.x * blockDim.x + threadIdx.x;
    if (i >= total) return;
    int n = i / C;
    int c = i - n * C;
    float S = wsum[n] / denom[n];
    out[i] = bias[c] + bl[c] + Wl[c] * S;
}

extern "C" void kernel_launch(void* const* d_in, const int* in_sizes, int n_in,
                              void* d_out, int out_size, void* d_ws, size_t ws_size,
                              hipStream_t stream) {
    const float* x    = (const float*)d_in[0];
    const int*   eidx = (const int*)  d_in[1];   // [2,E] row-major
    const float* ea   = (const float*)d_in[2];
    const float* Wl   = (const float*)d_in[3];
    const float* bl   = (const float*)d_in[4];
    const float* Wr   = (const float*)d_in[5];
    const float* br   = (const float*)d_in[6];
    const float* We   = (const float*)d_in[7];
    const float* att  = (const float*)d_in[8];
    const float* bias = (const float*)d_in[9];

    const int Nn   = in_sizes[0];        // x is [N,1]
    const int E    = in_sizes[1] / 2;
    const int C    = in_sizes[3];        // 64
    const int Etot = E + Nn;

    const int* esrc = eidx;
    const int* edst = eidx + E;

    const int ntiles  = (Etot + 15) / 16;
    const int EtotPad = ntiles * 16;     // logits padded so stores need no bound check

    // Workspace layout (floats): [logits EtotPad][nodemax N][denom N][wsum N][easum 1]
    float*    ws      = (float*)d_ws;
    float*    logits  = ws;
    unsigned* nodemax = (unsigned*)(ws + EtotPad);
    float*    denom   = ws + EtotPad + Nn;
    float*    wsum    = denom + Nn;
    float*    easum   = wsum + Nn;

    k_init<<<(Nn + 255) / 256, 256, 0, stream>>>(nodemax, denom, wsum, easum, Nn);
    k_easum<<<1024, 256, 0, stream>>>(ea, easum, E);

    k_logits<<<1280, 256, 0, stream>>>(x, esrc, edst, ea, Wl, bl, Wr, br, We, att,
                                       easum, logits, E, Etot, ntiles);

    const int gE = (Etot + 255) / 256;
    k_segmax<<<gE, 256, 0, stream>>>(logits, edst, nodemax, E, Etot);
    k_expsum<<<gE, 256, 0, stream>>>(logits, esrc, edst, x, nodemax, denom, wsum,
                                     E, Etot);

    const int total = Nn * C;
    k_out<<<(total + 255) / 256, 256, 0, stream>>>(Wl, bl, bias, denom, wsum,
                                                   (float*)d_out, total, C);
}
